// LearningNMS_23484881174582
// MI455X (gfx1250) — matrix operand order repaired and checked by reading, not compiled, not run on hardware
//
#include <hip/hip_runtime.h>
#include <hip/hip_bf16.h>

#define NN 768
#define FDIM 33
#define HID 128
#define PAIR_DIM 72
#define KPAD 96
#define THRESH 0.4f
#define EPSBN 1e-3f

typedef __attribute__((ext_vector_type(16))) __bf16 v16bf;
typedef __attribute__((ext_vector_type(8)))  float  v8f;

union Frag16 { v16bf v; uint4 q[2]; unsigned short u[16]; };

__device__ __forceinline__ unsigned short f2bf(float x) {
    union { float f; unsigned u; } v; v.f = x;
    unsigned r = v.u + 0x7FFFu + ((v.u >> 16) & 1u);
    return (unsigned short)(r >> 16);
}
__device__ __forceinline__ unsigned pack2(unsigned short a, unsigned short b) {
    return (unsigned)a | ((unsigned)b << 16);
}

__device__ __forceinline__ void pair_geom(const float* bx, int i, int j, float* g) {
    float xi = bx[i*4+0], yi = bx[i*4+1], wi = bx[i*4+2], hi = bx[i*4+3];
    float xj = bx[j*4+0], yj = bx[j*4+1], wj = bx[j*4+2], hj = bx[j*4+3];
    float ix1 = fmaxf(xi, xj), iy1 = fmaxf(yi, yj);
    float ix2 = fminf(xi + wi, xj + wj), iy2 = fminf(yi + hi, yj + hj);
    float inter = fmaxf(ix2 - ix1, 0.f) * fmaxf(iy2 - iy1, 0.f);
    float ai = wi * hi, aj = wj * hj;
    g[0] = inter / (ai + aj - inter + 1e-8f);
    g[1] = (xj - xi) / wi;
    g[2] = (yj - yi) / hi;
    g[3] = logf(wj / wi);
    g[4] = logf(hj / hi);
    g[5] = aj / (ai + 1e-8f);
}

// ---------------- K1: mask, row degrees, geom BN stats (per-row partials) --------
__global__ void k1_mask(const float* __restrict__ boxes, unsigned char* __restrict__ maskB,
                        float* __restrict__ deg, float* __restrict__ gsP) {
    __shared__ float red[256];
    const int i = blockIdx.x, t = threadIdx.x;
    float loc[13];
    #pragma unroll
    for (int v = 0; v < 13; ++v) loc[v] = 0.f;
    for (int j = t; j < NN; j += 256) {
        float g[6];
        pair_geom(boxes, i, j, g);
        float m = (g[0] > THRESH) ? 1.f : 0.f;
        maskB[i * NN + j] = (unsigned char)m;
        loc[0] += m;
        #pragma unroll
        for (int k = 0; k < 6; ++k) { loc[1 + k] += m * g[k]; loc[7 + k] += m * g[k] * g[k]; }
    }
    #pragma unroll
    for (int v = 0; v < 13; ++v) {
        red[t] = loc[v]; __syncthreads();
        for (int s = 128; s > 0; s >>= 1) { if (t < s) red[t] += red[t + s]; __syncthreads(); }
        if (t == 0) { gsP[i * 13 + v] = red[0]; if (v == 0) deg[i] = red[0]; }
        __syncthreads();
    }
}

// ---------------- K2: fold BN0 into layer-1 weights, bf16 transpose weights ------
__global__ void k2_prep(const float* __restrict__ bn0_g, const float* __restrict__ bn0_b,
                        const float* __restrict__ W1, const float* __restrict__ b1,
                        const float* __restrict__ W2,
                        const float* __restrict__ gsP, const float* __restrict__ deg,
                        const float* __restrict__ f,
                        unsigned short* __restrict__ w1t, float* __restrict__ b1f,
                        unsigned short* __restrict__ w2t, float* __restrict__ cntOut) {
    __shared__ float gstat[13];
    __shared__ float scale0[PAIR_DIM], shift0[PAIR_DIM];
    __shared__ float cntE;
    const int t = threadIdx.x;
    if (t < 13) { float s = 0.f; for (int b = 0; b < NN; ++b) s += gsP[b * 13 + t]; gstat[t] = s; }
    __syncthreads();
    if (t == 0) { cntE = gstat[0] + 1e-8f; cntOut[0] = cntE; }
    __syncthreads();
    if (t < PAIR_DIM) {
        float S, SS;
        if (t < 6) { S = gstat[1 + t]; SS = gstat[7 + t]; }
        else {
            int c = (t < 39) ? (t - 6) : (t - 39);   // mask symmetric: rowdeg == coldeg
            float s = 0.f, ss = 0.f;
            for (int i = 0; i < NN; ++i) {
                float w = deg[i], v = f[i * FDIM + c];
                s += w * v; ss += w * v * v;
            }
            S = s; SS = ss;
        }
        float mean = S / cntE;
        float var = SS / cntE - mean * mean; if (var < 0.f) var = 0.f;
        float sc = bn0_g[t] * rsqrtf(var + EPSBN);
        scale0[t] = sc; shift0[t] = bn0_b[t] - mean * sc;
    }
    __syncthreads();
    for (int idx = t; idx < HID * KPAD; idx += 256) {
        int n = idx / KPAD, k = idx - n * KPAD;
        float v = (k < PAIR_DIM) ? scale0[k] * W1[k * HID + n] : 0.f;
        w1t[idx] = f2bf(v);
    }
    for (int idx = t; idx < HID; idx += 256) {
        float s = b1[idx];
        for (int k = 0; k < PAIR_DIM; ++k) s += shift0[k] * W1[k * HID + idx];
        b1f[idx] = s;
    }
    for (int idx = t; idx < HID * HID; idx += 256) {
        int n = idx >> 7, k = idx & 127;
        w2t[idx] = f2bf(W2[k * HID + n]);
    }
}

// ---------------- K3: fused pairwise MLP via WMMA bf16 --------------------------
#define XS_STRIDE 104
#define HS_STRIDE 132
#define W1S_STRIDE 104
#define W2S_STRIDE 136
#define W1_OFF 0
#define W2_OFF 26624
#define XS_OFF 61440
#define HS_OFF 88064
#define ST_OFF 155648
#define BIAS_OFF 163840
#define SMEM_BYTES 164864
#define JCHUNK 192

__global__ void __launch_bounds__(256, 1)
k3_pair_mlp(const float* __restrict__ boxes, const unsigned char* __restrict__ maskB,
            const float* __restrict__ f,
            const unsigned short* __restrict__ w1t, const float* __restrict__ b1f,
            const unsigned short* __restrict__ w2t, const float* __restrict__ b2,
            float* __restrict__ pooledP, float* __restrict__ bn1P) {
    extern __shared__ char smem[];
    unsigned short* w1s = (unsigned short*)(smem + W1_OFF);
    unsigned short* w2s = (unsigned short*)(smem + W2_OFF);
    float* bias1 = (float*)(smem + BIAS_OFF);
    float* bias2 = (float*)(smem + BIAS_OFF + 512);
    const int tid = threadIdx.x;
    const int wv = tid >> 5, lane = tid & 31;
    const int nlane = lane & 15, half = lane >> 4;
    const int i0 = blockIdx.x * 16;
    const int irow = i0 + nlane;

    if (tid == 0) { __builtin_prefetch(f, 0, 1); __builtin_prefetch(w1t, 0, 1); }

    for (int idx = tid; idx < HID * KPAD; idx += 256) {
        int n = idx / KPAD, k = idx - n * KPAD;
        w1s[n * W1S_STRIDE + k] = w1t[idx];
    }
    for (int idx = tid; idx < HID * HID; idx += 256) {
        int n = idx >> 7, k = idx & 127;
        w2s[n * W2S_STRIDE + k] = w2t[idx];
    }
    if (tid < HID) { bias1[tid] = b1f[tid]; bias2[tid] = b2[tid]; }

    unsigned short* xs = (unsigned short*)(smem + XS_OFF) + wv * 16 * XS_STRIDE;
    float* hs = (float*)(smem + HS_OFF) + wv * 16 * HS_STRIDE;

    // j-invariant part of the x tile: f_i columns (6..38) and zero padding
    float fi32;
    {
        const float* fi = f + irow * FDIM;
        unsigned* dst = (unsigned*)(xs + nlane * XS_STRIDE) + half * 24;
        if (half == 0) {
            #pragma unroll
            for (int t = 0; t < 16; ++t) dst[3 + t] = pack2(f2bf(fi[2*t]), f2bf(fi[2*t+1]));
        } else {
            #pragma unroll
            for (int t = 12; t < 24; ++t) dst[t] = 0u;   // pad cols 72..95
        }
        fi32 = fi[32];
    }
    __syncthreads();

    float pool[8][8], s2[8];
    #pragma unroll
    for (int nt = 0; nt < 8; ++nt) {
        s2[nt] = 0.f;
        #pragma unroll
        for (int r = 0; r < 8; ++r) pool[nt][r] = 0.f;
    }

    const int jbase = blockIdx.y * JCHUNK;
    for (int j = jbase + wv; j < jbase + JCHUNK; j += 8) {
        float mfl[8]; float msum = 0.f;
        #pragma unroll
        for (int r = 0; r < 8; ++r) {
            mfl[r] = (float)maskB[(i0 + r + half * 8) * NN + j];
            msum += mfl[r];
        }
        if (!__any(msum > 0.f)) continue;   // wave-uniform column skip

        // ---- per-j part of x tile: geom cols 0..5 and f_j cols 39..71 ----
        {
            const float* fj = f + j * FDIM;
            unsigned* dst = (unsigned*)(xs + nlane * XS_STRIDE) + half * 24;
            if (half == 0) {
                float g[6]; pair_geom(boxes, irow, j, g);
                dst[0] = pack2(f2bf(g[0]), f2bf(g[1]));
                dst[1] = pack2(f2bf(g[2]), f2bf(g[3]));
                dst[2] = pack2(f2bf(g[4]), f2bf(g[5]));
                dst[19] = pack2(f2bf(fi32), f2bf(fj[0]));
                #pragma unroll
                for (int t = 0; t < 4; ++t) dst[20 + t] = pack2(f2bf(fj[1+2*t]), f2bf(fj[2+2*t]));
            } else {
                #pragma unroll
                for (int t = 0; t < 12; ++t) dst[t] = pack2(f2bf(fj[9+2*t]), f2bf(fj[10+2*t]));
            }
        }

        // ---- layer-1 A fragments (16x96 bf16, 3 k-steps) ----
        Frag16 a1[3];
        #pragma unroll
        for (int kt = 0; kt < 3; ++kt) {
            const char* base = (const char*)xs + nlane * (XS_STRIDE * 2) + (kt * 64 + half * 16);
            a1[kt].q[0] = *(const uint4*)(base);
            a1[kt].q[1] = *(const uint4*)(base + 32);
        }

        // ---- layer 1: h1 = relu(x @ W1' + b1') -> LDS (f32) ----
        #pragma unroll
        for (int nt = 0; nt < 8; ++nt) {
            v8f acc = {0.f,0.f,0.f,0.f,0.f,0.f,0.f,0.f};
            const int nch = nt * 16 + nlane;
            #pragma unroll
            for (int kt = 0; kt < 3; ++kt) {
                Frag16 bf;
                const char* wb = (const char*)w1s + nch * (W1S_STRIDE * 2) + (kt * 64 + half * 16);
                bf.q[0] = *(const uint4*)wb;
                bf.q[1] = *(const uint4*)(wb + 32);
                acc = __builtin_amdgcn_wmma_f32_16x16x32_bf16(false, a1[kt].v, false, bf.v,
                                                              (short)0, acc, false, false);
            }
            float bv = bias1[nch];
            #pragma unroll
            for (int r = 0; r < 8; ++r) {
                float h = fmaxf(acc[r] + bv, 0.f);
                hs[(r + half * 8) * HS_STRIDE + nt * 16 + nlane] = h;
            }
        }

        // ---- C->A transpose through LDS, cvt to bf16 (4 k-steps) ----
        Frag16 a2[4];
        #pragma unroll
        for (int kt = 0; kt < 4; ++kt) {
            const float* hb = hs + nlane * HS_STRIDE + kt * 32 + half * 8;
            float4 p0 = *(const float4*)(hb);
            float4 p1 = *(const float4*)(hb + 4);
            float4 p2 = *(const float4*)(hb + 16);
            float4 p3 = *(const float4*)(hb + 20);
            a2[kt].u[0] = f2bf(p0.x); a2[kt].u[1] = f2bf(p0.y); a2[kt].u[2] = f2bf(p0.z); a2[kt].u[3] = f2bf(p0.w);
            a2[kt].u[4] = f2bf(p1.x); a2[kt].u[5] = f2bf(p1.y); a2[kt].u[6] = f2bf(p1.z); a2[kt].u[7] = f2bf(p1.w);
            a2[kt].u[8]  = f2bf(p2.x); a2[kt].u[9]  = f2bf(p2.y); a2[kt].u[10] = f2bf(p2.z); a2[kt].u[11] = f2bf(p2.w);
            a2[kt].u[12] = f2bf(p3.x); a2[kt].u[13] = f2bf(p3.y); a2[kt].u[14] = f2bf(p3.z); a2[kt].u[15] = f2bf(p3.w);
        }

        // ---- layer 2 + masked pooling / BN1 moment-2 ----
        #pragma unroll
        for (int nt = 0; nt < 8; ++nt) {
            v8f acc = {0.f,0.f,0.f,0.f,0.f,0.f,0.f,0.f};
            const int nch = nt * 16 + nlane;
            #pragma unroll
            for (int kt = 0; kt < 4; ++kt) {
                Frag16 bf;
                const char* wb = (const char*)w2s + nch * (W2S_STRIDE * 2) + (kt * 64 + half * 16);
                bf.q[0] = *(const uint4*)wb;
                bf.q[1] = *(const uint4*)(wb + 32);
                acc = __builtin_amdgcn_wmma_f32_16x16x32_bf16(false, a2[kt].v, false, bf.v,
                                                              (short)0, acc, false, false);
            }
            float bv = bias2[nch];
            #pragma unroll
            for (int r = 0; r < 8; ++r) {
                float h = fmaxf(acc[r] + bv, 0.f);
                float mh = mfl[r] * h;
                pool[nt][r] += mh;
                s2[nt] += mh * h;
            }
        }
    }

    // ---- write pooled fragment to LDS for cross-wave reduction ----
    #pragma unroll
    for (int nt = 0; nt < 8; ++nt)
        #pragma unroll
        for (int r = 0; r < 8; ++r)
            hs[(r + half * 8) * HS_STRIDE + nt * 16 + nlane] = pool[nt][r];

    float* st = (float*)(smem + ST_OFF);
    #pragma unroll
    for (int nt = 0; nt < 8; ++nt) {
        // first moment is the row-sum of the pooled fragment: s1 = sum_r pool[nt][r]
        float s1 = pool[nt][0] + pool[nt][1] + pool[nt][2] + pool[nt][3]
                 + pool[nt][4] + pool[nt][5] + pool[nt][6] + pool[nt][7];
        float a = s1 + __shfl_xor(s1, 16, 32);
        float b = s2[nt] + __shfl_xor(s2[nt], 16, 32);
        if (half == 0) {
            st[wv * 256 + nt * 16 + nlane] = a;
            st[wv * 256 + 128 + nt * 16 + nlane] = b;
        }
    }
    __syncthreads();

    const int blk = blockIdx.x + gridDim.x * blockIdx.y;
    float* hsall = (float*)(smem + HS_OFF);
    for (int idx = tid; idx < 16 * HID; idx += 256) {
        int m = idx >> 7, c = idx & 127;
        float s = 0.f;
        #pragma unroll
        for (int w = 0; w < 8; ++w) s += hsall[w * 16 * HS_STRIDE + m * HS_STRIDE + c];
        pooledP[(blockIdx.y * NN + i0 + m) * HID + c] = s;
    }
    for (int idx = tid; idx < 256; idx += 256) {
        float s = 0.f;
        #pragma unroll
        for (int w = 0; w < 8; ++w) s += st[w * 256 + idx];
        bn1P[blk * 256 + idx] = s;
    }
}

// ---------------- K4a: finalize BN1 scale/shift ---------------------------------
__global__ void k4a_bn1(const float* __restrict__ bn1P, const float* __restrict__ cnt,
                        const float* __restrict__ g1, const float* __restrict__ b1n,
                        float* __restrict__ sc1, float* __restrict__ sh1) {
    const int c = threadIdx.x;
    float S = 0.f, SS = 0.f;
    for (int b = 0; b < 192; ++b) { S += bn1P[b * 256 + c]; SS += bn1P[b * 256 + 128 + c]; }
    float cntE = cnt[0];
    float mean = S / cntE;
    float var = SS / cntE - mean * mean; if (var < 0.f) var = 0.f;
    float sc = g1[c] * rsqrtf(var + EPSBN);
    sc1[c] = sc; sh1[c] = b1n[c] - mean * sc;
}

// ---------------- K4b: f_next = (s1*pooled/deg + t1) @ Wo + bo -------------------
__global__ void k4b_out(const float* __restrict__ pooledP, const float* __restrict__ deg,
                        const float* __restrict__ sc1, const float* __restrict__ sh1,
                        const float* __restrict__ Wo, const float* __restrict__ bo,
                        float* __restrict__ fout) {
    const int idx = blockIdx.x * 256 + threadIdx.x;
    if (idx >= NN * FDIM) return;
    const int i = idx / FDIM, o = idx - i * FDIM;
    const float inv = 1.f / (deg[i] + 1e-8f);
    float acc = bo[o];
    for (int c = 0; c < HID; ++c) {
        float p = pooledP[(0 * NN + i) * HID + c] + pooledP[(1 * NN + i) * HID + c]
                + pooledP[(2 * NN + i) * HID + c] + pooledP[(3 * NN + i) * HID + c];
        float v = sc1[c] * p * inv + sh1[c];
        acc += v * Wo[c * FDIM + o];
    }
    fout[idx] = acc;
}

// ---------------- K5: final MLP 33->128->128->1 + sigmoid ------------------------
__global__ void k5_final(const float* __restrict__ f,
                         const float* __restrict__ W1f, const float* __restrict__ b1f_,
                         const float* __restrict__ W2f, const float* __restrict__ b2f,
                         const float* __restrict__ W3f, const float* __restrict__ b3f,
                         float* __restrict__ out) {
    __shared__ float h[HID];
    __shared__ float red[HID];
    const int i = blockIdx.x, t = threadIdx.x;
    const float* fr = f + i * FDIM;
    float a = b1f_[t];
    for (int k = 0; k < FDIM; ++k) a += fr[k] * W1f[k * HID + t];
    h[t] = fmaxf(a, 0.f);
    __syncthreads();
    float a2 = b2f[t];
    for (int k = 0; k < HID; ++k) a2 += h[k] * W2f[k * HID + t];
    a2 = fmaxf(a2, 0.f);
    red[t] = a2 * W3f[t];
    __syncthreads();
    for (int s = 64; s > 0; s >>= 1) { if (t < s) red[t] += red[t + s]; __syncthreads(); }
    if (t == 0) { float z = red[0] + b3f[0]; out[i] = 1.f / (1.f + expf(-z)); }
}

// ---------------- host ----------------------------------------------------------
extern "C" void kernel_launch(void* const* d_in, const int* in_sizes, int n_in,
                              void* d_out, int out_size, void* d_ws, size_t ws_size,
                              hipStream_t stream) {
    (void)in_sizes; (void)n_in; (void)out_size; (void)ws_size;
    const float* interp = (const float*)d_in[0];
    const float* boxes  = (const float*)d_in[1];

    char* ws = (char*)d_ws;
    unsigned char* maskB = (unsigned char*)(ws + 0);          // 589824
    float* deg     = (float*)(ws + 589824);                    // 3072
    float* gsP     = (float*)(ws + 592896);                    // 39936
    float* cnt     = (float*)(ws + 632832);                    // 256
    unsigned short* w1t = (unsigned short*)(ws + 633088);      // 24576
    float* b1f     = (float*)(ws + 657664);                    // 512
    unsigned short* w2t = (unsigned short*)(ws + 658176);      // 32768
    float* bn1P    = (float*)(ws + 690944);                    // 192*256*4 = 196608
    float* sc1     = (float*)(ws + 887552);                    // 512
    float* sh1     = (float*)(ws + 888064);                    // 512
    float* fA      = (float*)(ws + 888576);                    // 101376
    float* fB      = (float*)(ws + 989952);                    // 101376
    float* pooledP = (float*)(ws + 1091328);                   // 4*768*128*4 = 1572864

    k1_mask<<<NN, 256, 0, stream>>>(boxes, maskB, deg, gsP);

    const float* fin = interp;
    float* fouts[2] = { fA, fB };
    for (int bi = 0; bi < 2; ++bi) {
        const int p = 2 + bi * 10;
        const float* bn0g = (const float*)d_in[p + 0];
        const float* bn0b = (const float*)d_in[p + 1];
        const float* W1   = (const float*)d_in[p + 2];
        const float* b1   = (const float*)d_in[p + 3];
        const float* W2   = (const float*)d_in[p + 4];
        const float* b2   = (const float*)d_in[p + 5];
        const float* bn1g = (const float*)d_in[p + 6];
        const float* bn1b = (const float*)d_in[p + 7];
        const float* Wo   = (const float*)d_in[p + 8];
        const float* bo   = (const float*)d_in[p + 9];

        k2_prep<<<1, 256, 0, stream>>>(bn0g, bn0b, W1, b1, W2, gsP, deg, fin,
                                       w1t, b1f, w2t, cnt);
        dim3 g3(48, 4);
        k3_pair_mlp<<<g3, 256, SMEM_BYTES, stream>>>(boxes, maskB, fin, w1t, b1f, w2t, b2,
                                                     pooledP, bn1P);
        k4a_bn1<<<1, 128, 0, stream>>>(bn1P, cnt, bn1g, bn1b, sc1, sh1);
        k4b_out<<<(NN * FDIM + 255) / 256, 256, 0, stream>>>(pooledP, deg, sc1, sh1, Wo, bo,
                                                             fouts[bi]);
        fin = fouts[bi];
    }

    k5_final<<<NN, HID, 0, stream>>>(fin,
                                     (const float*)d_in[22], (const float*)d_in[23],
                                     (const float*)d_in[24], (const float*)d_in[25],
                                     (const float*)d_in[26], (const float*)d_in[27],
                                     (float*)d_out);
}
